// MiddleLayerDecoderSplit_42820823941127
// MI455X (gfx1250) — matrix operand order repaired and checked by reading, not compiled, not run on hardware
//
#include <hip/hip_runtime.h>
#include <stdint.h>

#define N_PARENTS 4096
#define KPTS      32
#define NEIGH     128
#define FEAT      256
#define HD1       512
#define HD2       512
#define HD3       256
#define NROWS     (N_PARENTS * KPTS)      // 131072

// LDS layout for stage_b (bytes): 2 weight buffers + h1 + h2 = 256 KB <= 320 KB/WGP
#define SW0_OFF   0
#define SW1_OFF   65536
#define SH1_OFF   131072
#define SH2_OFF   196608
#define LDS_BYTES 262144

// ROCm 7.2 (clang-22) declares the 5-arg tensor_load_to_lds builtin; the
// amdgpu-toolchain (clang-23) uses a 6-arg form -> fall back to async loads.
#if !defined(__clang_major__) || __clang_major__ < 23
#define USE_TDM 1
#else
#define USE_TDM 0
#endif

typedef __bf16 v16bf __attribute__((ext_vector_type(16)));
typedef __bf16 v8bf  __attribute__((ext_vector_type(8)));
typedef float  v8f   __attribute__((ext_vector_type(8)));

static __device__ __forceinline__ unsigned short f2bf(float f) {
    union { float f; unsigned u; } v; v.f = f;
    unsigned r = v.u + 0x7FFFu + ((v.u >> 16) & 1u);   // round-to-nearest-even
    return (unsigned short)(r >> 16);
}

static __device__ __forceinline__ v16bf cat8(v8bf lo, v8bf hi) {
    return __builtin_shufflevector(lo, hi, 0,1,2,3,4,5,6,7,8,9,10,11,12,13,14,15);
}

#if USE_TDM
typedef unsigned int tdm_v4u __attribute__((ext_vector_type(4)));
typedef int          tdm_v8i __attribute__((ext_vector_type(8)));
typedef int          tdm_v4i __attribute__((ext_vector_type(4)));

// One TDM descriptor moves a full 64 KB weight tile (8192 x 8B elements, 1-D)
// from global to LDS.  D# bitfields per CDNA5 ISA §8.3-8.4:
//   group0: count=1 | lds_addr | global_addr[56:0] | type=2
//   group1: data_size=3 (8B), tensor_dim0=tile_dim0=8192, dim0_stride=8192,
//           workgroup_mask=0 (not in a cluster), no pad/iterate/gather.
static __device__ __forceinline__ void tdm_load_tile(
        const unsigned short* gsrc, unsigned short* ldst) {
    unsigned long long ga = (unsigned long long)(uintptr_t)gsrc;
    unsigned la = (unsigned)(uintptr_t)(void*)ldst;     // LDS byte address
    tdm_v4u g0;
    g0[0] = 1u;                                          // count=1 (valid user D#)
    g0[1] = la;                                          // lds_addr
    g0[2] = (unsigned)ga;                                // global_addr[31:0]
    g0[3] = (unsigned)((ga >> 32) & 0x01FFFFFFu)         // global_addr[56:32]
          | 0x80000000u;                                 // type=2 ("image")
    tdm_v8i g1;
    g1[0] = 0x00030000;                                  // data_size=8B
    g1[1] = (int)(8192u << 16);                          // tensor_dim0[15:0]
    g1[2] = 0;                                           // tensor_dim0 hi, dim1 lo
    g1[3] = (int)(8192u << 16);                          // tile_dim0 = 8192
    g1[4] = 0;                                           // tile_dim1/2 unused (1-D)
    g1[5] = 8192;                                        // tensor_dim0_stride lo
    g1[6] = 0;
    g1[7] = 0;
    tdm_v4i gz = {0, 0, 0, 0};                           // groups 2/3: <=2-D tensor
    __builtin_amdgcn_tensor_load_to_lds(g0, g1, gz, gz, 0);
}
#else
// Fallback: 64 KB tile as 32 per-wave async-to-LDS ops; ASYNCcnt completes in
// order, so "s_wait_asynccnt 32" guarantees the previous tile is resident.
static __device__ __forceinline__ void issue_async_tile(
        const unsigned short* __restrict__ gsrc, unsigned short* ldst, int tid) {
    #pragma unroll 4
    for (int e = tid; e < 4096; e += 128) {              // 4096 x 16B = 64 KB
        unsigned laddr = (unsigned)(uintptr_t)(void*)(ldst + e * 8);
        const unsigned short* g = gsrc + e * 8;
        asm volatile("global_load_async_to_lds_b128 %0, %1, off"
                     :: "v"(laddr), "v"(g) : "memory");
    }
}
#endif

// ---------------------------------------------------------------------------
// Kernel 1: transpose W2 (512x512) and W3 (512x256) into N-major bf16 so that
// each WMMA B-fragment is two contiguous 16B LDS loads per lane.
// ---------------------------------------------------------------------------
__global__ void prep_weights(const float* __restrict__ W2,
                             const float* __restrict__ W3,
                             unsigned short* __restrict__ w2t,   // [512][512]
                             unsigned short* __restrict__ w3t) { // [256][512]
    int idx = blockIdx.x * blockDim.x + threadIdx.x;
    if (idx < HD2 * HD1) {
        int n = idx >> 9, k = idx & 511;
        w2t[idx] = f2bf(W2[k * HD2 + n]);
    } else if (idx < HD2 * HD1 + HD3 * HD2) {
        int j = idx - HD2 * HD1;
        int n = j >> 9, k = j & 511;
        w3t[j] = f2bf(W3[k * HD3 + n]);
    }
}

// ---------------------------------------------------------------------------
// Kernel 2: per-parent work.  Decoder (128->96) -> rel points; per-parent
// feature half of layer 1: f1base = features @ W1[0:256] + b1 (shared by all
// 32 rows of the cluster -> 32x less layer-1 work).  Also emits cluster ids.
// ---------------------------------------------------------------------------
__global__ void __launch_bounds__(128) stage_a(
        const float* __restrict__ in,     // (4096, 384)
        const float* __restrict__ Wdec,   // (128, 96)
        const float* __restrict__ bdec,   // (96)
        const float* __restrict__ W1,     // (259, 512)
        const float* __restrict__ b1,     // (512)
        float* __restrict__ f1base,       // ws (4096, 512)
        float* __restrict__ relws,        // ws (131072, 3)
        float* __restrict__ out)          // d_out
{
    __shared__ float sIn[NEIGH + FEAT];
    const int p   = blockIdx.x;
    const int tid = threadIdx.x;
    const float* row = in + (size_t)p * (NEIGH + FEAT);
    for (int e = tid; e < NEIGH + FEAT; e += 128) sIn[e] = row[e];
    __syncthreads();

    // relative points (96 per parent)
    if (tid < 96) {
        float acc = bdec[tid];
        #pragma unroll 8
        for (int k = 0; k < NEIGH; ++k) acc += sIn[k] * Wdec[k * 96 + tid];
        relws[p * 96 + tid] = acc;
        out[p * 96 + tid]   = acc;                 // output 0: rel_points
    }
    // output 2: cluster ids (int32 promoted to f32 by the tuple concat)
    if (tid < KPTS) {
        out[(size_t)NROWS * 3 + (size_t)NROWS * HD3 + (size_t)p * KPTS + tid] = (float)p;
    }

    // f1base: 512 outputs, 4 per thread
    float a0 = b1[tid], a1 = b1[tid + 128], a2 = b1[tid + 256], a3 = b1[tid + 384];
    for (int k = 0; k < FEAT; ++k) {
        float fv = sIn[NEIGH + k];
        const float* wr = W1 + k * HD1;
        a0 += fv * wr[tid];
        a1 += fv * wr[tid + 128];
        a2 += fv * wr[tid + 256];
        a3 += fv * wr[tid + 384];
    }
    float* dst = f1base + (size_t)p * HD1;
    dst[tid] = a0; dst[tid + 128] = a1; dst[tid + 256] = a2; dst[tid + 384] = a3;
}

// ---------------------------------------------------------------------------
// Kernel 3: fused layers 1(rel part)+2+3 with BF16 WMMA.
// One block = 4 waves = 64 rows (2 parents).  Weight tiles (64 N-rows x 512 K,
// 64 KB) are double-buffered via the Tensor Data Mover (one tensor_load_to_lds
// per tile, TENSORcnt-tracked) so the fetch of chunk c+1 overlaps the WMMA
// work of chunk c; layers 2 and 3 form one 12-chunk pipeline (8 of W2^T + 4 of
// W3^T) with seamless ping-pong across the layer boundary.
// ---------------------------------------------------------------------------
__global__ void __launch_bounds__(128) stage_b(
        const float* __restrict__ f1base,          // (4096, 512)
        const float* __restrict__ relws,           // (131072, 3)
        const float* __restrict__ W1,              // rel rows at +256*512
        const float* __restrict__ b2,              // (512)
        const float* __restrict__ b3,              // (256)
        const unsigned short* __restrict__ w2t,    // (512, 512) bf16 N-major
        const unsigned short* __restrict__ w3t,    // (256, 512) bf16 N-major
        float* __restrict__ out)
{
    extern __shared__ char smem[];
    unsigned short* sW0 = (unsigned short*)(smem + SW0_OFF);
    unsigned short* sW1 = (unsigned short*)(smem + SW1_OFF);
    unsigned short* sH1 = (unsigned short*)(smem + SH1_OFF);   // 64 x 512 bf16
    unsigned short* sH2 = (unsigned short*)(smem + SH2_OFF);   // 64 x 512 bf16

    const int tid     = threadIdx.x;
    const int wave    = tid >> 5;
    const int lane    = tid & 31;
    const int half    = lane >> 4;      // lane half selects K sub-range (ISA layout)
    const int l16     = lane & 15;
    const int R0      = wave * 16;      // wave's 16 rows within the 64-row tile
    const int rowBase = blockIdx.x * 64;

    const float* W1r = W1 + 256 * HD1;  // (3, 512) rel-coordinate rows of W1

    // Prefetch weight chunk 0 (W2^T rows 0..63) while phase 1 runs on the VALUs.
#if USE_TDM
    if (wave == 0) tdm_load_tile(w2t, sW0);
#else
    issue_async_tile(w2t, sW0, tid);
#endif

    // ---- Phase 1: h1 = relu(f1base[parent] + rel . W1r) -> sH1 (bf16) ----
    // Stage the block's f1 rows (2x512 f32), rel points (64x3 f32) and the
    // rel-weight block W1r (3x512 f32) in LDS; temps live in the sH2 region,
    // which is not written until the first layer-2 epilogue.
    float* tF1  = (float*)(smem + SH2_OFF);                 // [2][512]
    float* tRel = (float*)(smem + SH2_OFF + 4096);          // [64][3]
    float* tW1r = (float*)(smem + SH2_OFF + 4096 + 768);    // [3][512]
    for (int e = tid; e < 1024; e += 128) tF1[e]  = f1base[(size_t)(rowBase >> 5) * HD1 + e];
    for (int e = tid; e < 192; e += 128)  tRel[e] = relws[(size_t)rowBase * 3 + e];
    for (int e = tid; e < 1536; e += 128) tW1r[e] = W1r[e];
    __syncthreads();

    for (int g = tid; g < 4096; g += 128) {          // 8 packed bf16 per store
        int r  = g >> 6;                             // row in tile 0..63
        int c0 = (g & 63) * 8;
        const float* fb = tF1 + ((r >> 5) & 1) * 512;
        float rx = tRel[r * 3], ry = tRel[r * 3 + 1], rz = tRel[r * 3 + 2];
        unsigned wds[4];
        #pragma unroll
        for (int j = 0; j < 8; j += 2) {
            float v0 = fb[c0 + j]     + rx * tW1r[c0 + j]
                     + ry * tW1r[HD1 + c0 + j]     + rz * tW1r[2 * HD1 + c0 + j];
            float v1 = fb[c0 + j + 1] + rx * tW1r[c0 + j + 1]
                     + ry * tW1r[HD1 + c0 + j + 1] + rz * tW1r[2 * HD1 + c0 + j + 1];
            v0 = v0 > 0.f ? v0 : 0.f;
            v1 = v1 > 0.f ? v1 : 0.f;
            wds[j >> 1] = (unsigned)f2bf(v0) | ((unsigned)f2bf(v1) << 16);
        }
        *(uint4*)(sH1 + (size_t)g * 8) = make_uint4(wds[0], wds[1], wds[2], wds[3]);
    }
    __syncthreads();

    // ---- Unified 12-chunk WMMA pipeline ----
    // chunks 0..7 : h2 = relu(h1 @ W2 + b2) -> sH2 (bf16)
    // chunks 8..11: h3 = relu(h2 @ W3 + b3) -> global f32
    const unsigned short* aRow1 = sH1 + (R0 + l16) * 512;
    const unsigned short* aRow2 = sH2 + (R0 + l16) * 512;
    float* hout = out + (size_t)NROWS * 3;           // output 1 section

    for (int c = 0; c < 12; ++c) {
        // Prefetch chunk c+1 into the other buffer (last read at chunk c-1,
        // protected by the trailing barrier of the previous iteration).
        // In-order completion per wave: waiting cnt<=1 (TDM) / <=32 (async)
        // guarantees chunk c is fully resident before the barrier releases.
        if (c + 1 < 12) {
            const unsigned short* nsrc = (c + 1 < 8)
                ? (w2t + (size_t)(c + 1) * 64 * 512)
                : (w3t + (size_t)(c + 1 - 8) * 64 * 512);
            unsigned short* ndst = ((c + 1) & 1) ? sW1 : sW0;
#if USE_TDM
            if (wave == 0) {
                tdm_load_tile(nsrc, ndst);
                __builtin_amdgcn_s_wait_tensorcnt(1);
            }
#else
            issue_async_tile(nsrc, ndst, tid);
            asm volatile("s_wait_asynccnt 32" ::: "memory");
#endif
        } else {
#if USE_TDM
            if (wave == 0) __builtin_amdgcn_s_wait_tensorcnt(0);
#else
            asm volatile("s_wait_asynccnt 0" ::: "memory");
#endif
        }
        __syncthreads();

        const unsigned short* sW   = (c & 1) ? sW1 : sW0;
        const unsigned short* aRow = (c < 8) ? aRow1 : aRow2;

        v8f cc[4] = {};
        for (int k0 = 0; k0 < 512; k0 += 32) {
            // A fragment (16x32 bf16): lane<16 K0..7|16..23, lane>=16 K8..15|24..31
            v16bf a = cat8(*(const v8bf*)(aRow + k0 + half * 8),
                           *(const v8bf*)(aRow + k0 + 16 + half * 8));
            // B fragments: N-major rows, lanes<16 K0..7|16..23, lanes>=16 K8..15|24..31
            const unsigned short* bb = sW + l16 * 512 + k0 + half * 8;
            v16bf b0v = cat8(*(const v8bf*)(bb),            *(const v8bf*)(bb + 16));
            v16bf b1v = cat8(*(const v8bf*)(bb + 16 * 512), *(const v8bf*)(bb + 16 * 512 + 16));
            v16bf b2v = cat8(*(const v8bf*)(bb + 32 * 512), *(const v8bf*)(bb + 32 * 512 + 16));
            v16bf b3v = cat8(*(const v8bf*)(bb + 48 * 512), *(const v8bf*)(bb + 48 * 512 + 16));
            cc[0] = __builtin_amdgcn_wmma_f32_16x16x32_bf16(false, a, false, b0v, (short)0, cc[0], false, false);
            cc[1] = __builtin_amdgcn_wmma_f32_16x16x32_bf16(false, a, false, b1v, (short)0, cc[1], false, false);
            cc[2] = __builtin_amdgcn_wmma_f32_16x16x32_bf16(false, a, false, b2v, (short)0, cc[2], false, false);
            cc[3] = __builtin_amdgcn_wmma_f32_16x16x32_bf16(false, a, false, b3v, (short)0, cc[3], false, false);
        }

        if (c < 8) {
            #pragma unroll
            for (int t = 0; t < 4; ++t) {           // +bias, relu, bf16 -> sH2
                int colAbs = c * 64 + t * 16 + l16;
                float bias = b2[colAbs];
                #pragma unroll
                for (int v = 0; v < 8; ++v) {       // D: lane<16 M=v, lane>=16 M=v+8
                    float x = cc[t][v] + bias;
                    x = x > 0.f ? x : 0.f;
                    sH2[(R0 + half * 8 + v) * 512 + colAbs] = f2bf(x);
                }
            }
        } else {
            #pragma unroll
            for (int t = 0; t < 4; ++t) {           // +bias, relu -> global f32
                int colAbs = (c - 8) * 64 + t * 16 + l16;
                float bias = b3[colAbs];
                #pragma unroll
                for (int v = 0; v < 8; ++v) {
                    float x = cc[t][v] + bias;
                    x = x > 0.f ? x : 0.f;
                    int grow = rowBase + R0 + half * 8 + v;
                    hout[(size_t)grow * HD3 + colAbs] = x;
                }
            }
        }
        __syncthreads();
    }
}

// ---------------------------------------------------------------------------
extern "C" void kernel_launch(void* const* d_in, const int* in_sizes, int n_in,
                              void* d_out, int out_size, void* d_ws, size_t ws_size,
                              hipStream_t stream) {
    (void)in_sizes; (void)n_in; (void)out_size; (void)ws_size;
    const float* in_feat = (const float*)d_in[0];
    const float* Wdec    = (const float*)d_in[1];
    const float* bdec    = (const float*)d_in[2];
    const float* W1      = (const float*)d_in[3];
    const float* b1      = (const float*)d_in[4];
    const float* W2      = (const float*)d_in[5];
    const float* b2      = (const float*)d_in[6];
    const float* W3      = (const float*)d_in[7];
    const float* b3      = (const float*)d_in[8];
    float* out = (float*)d_out;

    char* ws = (char*)d_ws;
    float*          f1base = (float*)ws;                                   // 8 MB
    float*          relws  = (float*)(ws + 8388608);                       // 1.5 MB
    unsigned short* w2t    = (unsigned short*)(ws + 8388608 + 1572864);    // 512 KB
    unsigned short* w3t    = (unsigned short*)(ws + 8388608 + 1572864 + 524288); // 256 KB

    prep_weights<<<(HD2 * HD1 + HD3 * HD2 + 255) / 256, 256, 0, stream>>>(W2, W3, w2t, w3t);
    stage_a<<<N_PARENTS, 128, 0, stream>>>(in_feat, Wdec, bdec, W1, b1, f1base, relws, out);
    stage_b<<<NROWS / 64, 128, LDS_BYTES, stream>>>(f1base, relws, W1, b2, b3, w2t, w3t, out);
}
